// Attention_Layer_17781164605945
// MI455X (gfx1250) — matrix operand order repaired and checked by reading, not compiled, not run on hardware
//
#include <hip/hip_runtime.h>
#include <hip/hip_bf16.h>

typedef __attribute__((ext_vector_type(16))) __bf16        v16bf;
typedef __attribute__((ext_vector_type(8)))  float         v8f;
typedef __attribute__((ext_vector_type(4)))  unsigned int  v4u;
typedef __attribute__((ext_vector_type(4)))  float         v4f;

#define NB    8
#define TEL   4096
#define TDL   1024
#define HDIM  256
#define NHID  256
#define TM    64        // d rows per workgroup
#define TNK   64        // e rows per key tile
#define NWAVE 4
#define NTHR  128
#define RS    264       // LDS row stride (halves) for 256-wide rows (pad +8)
#define ES    72        // LDS row stride for e_t rows (64 wide, pad +8)
#define PS    72        // LDS row stride for P rows (64 wide)
#define WSR   520       // LDS row stride for W tile rows (512 wide)

static __device__ __forceinline__ unsigned short f2bf(float f) {
  unsigned int u = __builtin_bit_cast(unsigned int, f);
  u += 0x7FFFu + ((u >> 16) & 1u);             // round-to-nearest-even
  return (unsigned short)(u >> 16);
}

// Build a v16bf WMMA fragment from two 16-byte chunks (per documented
// 16-bit A/B VGPR layouts: halves [0..7] then [8..15]).
static __device__ __forceinline__ v16bf frag2(const unsigned short* p0,
                                              const unsigned short* p1) {
  union { v4u u[2]; v16bf b; } t;
  t.u[0] = *(const v4u*)p0;
  t.u[1] = *(const v4u*)p1;
  return t.b;
}

__global__ __launch_bounds__(NTHR, 1)
void attn_fused_kernel(const float* __restrict__ e, const float* __restrict__ d,
                       const float* __restrict__ W, const float* __restrict__ bias,
                       float* __restrict__ out) {
  __shared__ unsigned short d_bf[TM * RS];          // 33,792 B
  __shared__ unsigned short e_bf[TNK * RS];         // 33,792 B (reused as W tile)
  __shared__ unsigned short e_t [HDIM * ES];        // 36,864 B (reused as V)
  __shared__ unsigned short p_bf[NWAVE * 16 * PS];  //  9,216 B

  const int tid  = threadIdx.x;
  const int wave = tid >> 5;
  const int lane = tid & 31;
  const int ln   = lane & 15;     // column / M-row within fragment
  const int hi   = lane >> 4;     // lane half selects K / M+8 split
  const int m0   = wave * 16;     // this wave's row offset in the 64-row tile

  const int wg    = blockIdx.x;
  const int batch = wg >> 4;                // TD/TM == 16
  const int mbase = (wg & 15) * TM;

  const float* dptr = d + ((size_t)batch * TDL + mbase) * HDIM;
  const float* eptr = e + (size_t)batch * TEL * HDIM;

  // ---- stage d tile (64 x 256) as bf16, row-major ----
  #pragma unroll
  for (int i = 0; i < 32; ++i) {
    int idx = tid + i * NTHR;               // float4 index 0..4095
    int row = idx >> 6;
    int c4  = idx & 63;
    v4f v = *(const v4f*)(dptr + row * HDIM + c4 * 4);
    unsigned short* s = &d_bf[row * RS + c4 * 4];
    s[0] = f2bf(v[0]); s[1] = f2bf(v[1]); s[2] = f2bf(v[2]); s[3] = f2bf(v[3]);
  }

  const v8f zf = {0.f,0.f,0.f,0.f,0.f,0.f,0.f,0.f};
  v8f acc[16];                               // V accumulator: 16 rows x 256 cols
  #pragma unroll
  for (int i = 0; i < 16; ++i) acc[i] = zf;
  float mrow[8], lrow[8];
  #pragma unroll
  for (int r = 0; r < 8; ++r) { mrow[r] = -3.0e38f; lrow[r] = 0.f; }

  for (int jt = 0; jt < TEL / TNK; ++jt) {
    const float* ep = eptr + (size_t)jt * TNK * HDIM;
    __syncthreads();                         // prev tile fully consumed
    #pragma unroll
    for (int i = 0; i < 32; ++i) {
      int idx = tid + i * NTHR;
      int row = idx >> 6;
      int c4  = idx & 63;
      v4f v = *(const v4f*)(ep + row * HDIM + c4 * 4);
      unsigned short b0 = f2bf(v[0]), b1 = f2bf(v[1]);
      unsigned short b2 = f2bf(v[2]), b3 = f2bf(v[3]);
      unsigned short* s = &e_bf[row * RS + c4 * 4];
      s[0] = b0; s[1] = b1; s[2] = b2; s[3] = b3;
      int c = c4 * 4;                        // transposed copy for the PV matmul
      e_t[(c + 0) * ES + row] = b0;
      e_t[(c + 1) * ES + row] = b1;
      e_t[(c + 2) * ES + row] = b2;
      e_t[(c + 3) * ES + row] = b3;
    }
    if (jt + 1 < TEL / TNK) {               // hint next tile toward the caches
      const float* np = eptr + (size_t)(jt + 1) * TNK * HDIM;
      #pragma unroll
      for (int i = 0; i < 4; ++i)
        __builtin_prefetch(np + (size_t)(tid + i * NTHR) * 32, 0, 1);
    }
    __syncthreads();

    // ---- S(16x64) = d_rows x e_tile^T, K = 256 ----
    v8f sv[4] = {zf, zf, zf, zf};
    #pragma unroll
    for (int kt = 0; kt < 8; ++kt) {
      int kb = kt * 32 + hi * 8;
      v16bf a = frag2(&d_bf[(m0 + ln) * RS + kb], &d_bf[(m0 + ln) * RS + kb + 16]);
      #pragma unroll
      for (int nt = 0; nt < 4; ++nt) {
        const unsigned short* bp = &e_bf[(nt * 16 + ln) * RS + kt * 32 + hi * 8];
        v16bf bb = frag2(bp, bp + 16);
        sv[nt] = __builtin_amdgcn_wmma_f32_16x16x32_bf16(
            false, a, false, bb, (short)0, sv[nt], false, false);
      }
    }

    // ---- online softmax over this key tile ----
    float pm[8];
    #pragma unroll
    for (int r = 0; r < 8; ++r)
      pm[r] = fmaxf(fmaxf(sv[0][r], sv[1][r]), fmaxf(sv[2][r], sv[3][r]));
    #pragma unroll
    for (int mk = 1; mk <= 8; mk <<= 1) {
      #pragma unroll
      for (int r = 0; r < 8; ++r)
        pm[r] = fmaxf(pm[r], __shfl_xor(pm[r], mk, 32));
    }
    float fr[8];
    #pragma unroll
    for (int r = 0; r < 8; ++r) {
      float mn = fmaxf(mrow[r], pm[r]);
      fr[r]   = __expf(mrow[r] - mn);
      mrow[r] = mn;
      lrow[r] *= fr[r];
    }
    #pragma unroll
    for (int nt = 0; nt < 16; ++nt) {
      #pragma unroll
      for (int r = 0; r < 8; ++r) acc[nt][r] *= fr[r];
    }
    #pragma unroll
    for (int nt = 0; nt < 4; ++nt) {
      #pragma unroll
      for (int r = 0; r < 8; ++r) {
        float p = __expf(sv[nt][r] - mrow[r]);
        lrow[r] += p;
        p_bf[(wave * 16 + r + 8 * hi) * PS + nt * 16 + ln] = f2bf(p);
      }
    }

    // ---- acc(16x256) += P(16x64) x e_tile(64x256) ----
    #pragma unroll
    for (int kt = 0; kt < 2; ++kt) {
      int kb = kt * 32 + hi * 8;
      v16bf a = frag2(&p_bf[(wave * 16 + ln) * PS + kb],
                      &p_bf[(wave * 16 + ln) * PS + kb + 16]);
      #pragma unroll
      for (int nt = 0; nt < 16; ++nt) {
        const unsigned short* bp = &e_t[(nt * 16 + ln) * ES + kt * 32 + hi * 8];
        v16bf bb = frag2(bp, bp + 16);
        acc[nt] = __builtin_amdgcn_wmma_f32_16x16x32_bf16(
            false, a, false, bb, (short)0, acc[nt], false, false);
      }
    }
  }

  // ================= epilogue: tanh([V, d] @ W + b) =================
  __syncthreads();                           // everyone done with e_bf / e_t
  unsigned short* v_bf = e_t;                // 4*16*RS = 16,896 halves <= 18,432

  #pragma unroll
  for (int mk = 1; mk <= 8; mk <<= 1) {      // finish row sums across lanes
    #pragma unroll
    for (int r = 0; r < 8; ++r) lrow[r] += __shfl_xor(lrow[r], mk, 32);
  }
  float rl[8];
  #pragma unroll
  for (int r = 0; r < 8; ++r) rl[r] = 1.0f / lrow[r];
  #pragma unroll
  for (int nt = 0; nt < 16; ++nt) {          // normalized V -> LDS bf16; acc dies
    #pragma unroll
    for (int r = 0; r < 8; ++r)
      v_bf[(wave * 16 + r + 8 * hi) * RS + nt * 16 + ln] = f2bf(acc[nt][r] * rl[r]);
  }

  unsigned short* wtile = e_bf;              // 16*WSR = 8,320 halves <= 16,896
  const int mg = batch * TDL + mbase + m0;
  for (int nt = 0; nt < 16; ++nt) {
    __syncthreads();
    // W[:, nt*16 .. nt*16+16) transposed: wtile[nl][k] = W[k][nt*16+nl], bf16
    #pragma unroll
    for (int i = 0; i < 64; ++i) {
      int idx = tid + i * NTHR;              // 0..8191
      int k   = idx >> 4;
      int nl  = idx & 15;
      wtile[nl * WSR + k] = f2bf(W[k * NHID + nt * 16 + nl]);
    }
    __syncthreads();
    float bv = bias[nt * 16 + ln];
    v8f o = {bv, bv, bv, bv, bv, bv, bv, bv};
    #pragma unroll
    for (int kt = 0; kt < 8; ++kt) {         // K 0..255 : value half
      int kb = kt * 32 + hi * 8;
      v16bf a = frag2(&v_bf[(wave * 16 + ln) * RS + kb],
                      &v_bf[(wave * 16 + ln) * RS + kb + 16]);
      const unsigned short* bp = &wtile[ln * WSR + kt * 32 + hi * 8];
      v16bf bb = frag2(bp, bp + 16);
      o = __builtin_amdgcn_wmma_f32_16x16x32_bf16(
          false, a, false, bb, (short)0, o, false, false);
    }
    #pragma unroll
    for (int kt = 0; kt < 8; ++kt) {         // K 256..511 : d half
      int kb = kt * 32 + hi * 8;
      v16bf a = frag2(&d_bf[(m0 + ln) * RS + kb],
                      &d_bf[(m0 + ln) * RS + kb + 16]);
      const unsigned short* bp = &wtile[ln * WSR + 256 + kt * 32 + hi * 8];
      v16bf bb = frag2(bp, bp + 16);
      o = __builtin_amdgcn_wmma_f32_16x16x32_bf16(
          false, a, false, bb, (short)0, o, false, false);
    }
    #pragma unroll
    for (int r = 0; r < 8; ++r)
      out[(size_t)(mg + r + 8 * hi) * NHID + nt * 16 + ln] = tanhf(o[r]);
  }
}

extern "C" void kernel_launch(void* const* d_in, const int* in_sizes, int n_in,
                              void* d_out, int out_size, void* d_ws, size_t ws_size,
                              hipStream_t stream) {
  (void)in_sizes; (void)n_in; (void)out_size; (void)d_ws; (void)ws_size;
  const float* e    = (const float*)d_in[0];
  const float* d    = (const float*)d_in[1];
  const float* W    = (const float*)d_in[2];
  const float* bias = (const float*)d_in[3];
  float* out = (float*)d_out;
  dim3 grid(NB * (TDL / TM));   // 8 * 16 = 128 workgroups
  attn_fused_kernel<<<grid, NTHR, 0, stream>>>(e, d, W, bias, out);
}
